// MultiHeadDotProductAttention_10986526343956
// MI455X (gfx1250) — matrix operand order repaired and checked by reading, not compiled, not run on hardware
//
#include <hip/hip_runtime.h>

// ---------------------------------------------------------------------------
// MHA forward for B=2, L=2048, C=1024, H=16, D=64 on gfx1250 (wave32, WMMA)
// ---------------------------------------------------------------------------

typedef __attribute__((ext_vector_type(16))) _Float16 v16h;
typedef __attribute__((ext_vector_type(8)))  _Float16 v8h;
typedef __attribute__((ext_vector_type(2)))  _Float16 v2h;
typedef __attribute__((ext_vector_type(8)))  float    v8f;
typedef unsigned int u32;
typedef __attribute__((ext_vector_type(4)))  u32 u32x4;
typedef __attribute__((ext_vector_type(8)))  u32 u32x8;

union AFrag {
    v16h v;
    v8h  h[2];
};

static constexpr int kC  = 1024;
static constexpr int kL  = 2048;
static constexpr int kB  = 2;
static constexpr int kH  = 16;
static constexpr int kD  = 64;
static constexpr int kNT = kB * kL;   // 4096 token rows

// ---------------------------------------------------------------------------
// CDNA5 async global->LDS copy (ASYNCcnt) and TDM tensor load (TENSORcnt)
// per CDNA5 ISA
// ---------------------------------------------------------------------------
__device__ __forceinline__ void async_load_b128_to_lds(unsigned lds_off,
                                                       const void* gptr) {
    asm volatile("global_load_async_to_lds_b128 %0, %1, off"
                 :: "v"(lds_off), "v"(gptr) : "memory");
}
__device__ __forceinline__ void wait_asynccnt0() {
    asm volatile("s_wait_asynccnt 0x0" ::: "memory");
}

// TDM: DMA a 32-row x 64-elem f16 tile (row stride 64 elems) into LDS.
// D# group0: {count=1 | lds_addr | global_addr[56:0] | type=2}
// D# group1: {data_size=1 (2B), tensor_dim0=64, tensor_dim1=2048,
//             tile_dim0=64, tile_dim1=32, tensor_dim0_stride=64}
__device__ __forceinline__ void tdm_load_v_tile_to_lds(unsigned lds_addr,
                                                       const void* gaddr) {
    unsigned long long ga = (unsigned long long)(uintptr_t)gaddr;
    u32x4 g0 = { 1u,                                   // count=1, user mode
                 lds_addr,                             // lds_addr (bytes)
                 (u32)ga,                              // global_addr[31:0]
                 ((u32)(ga >> 32) & 0x01FFFFFFu) | 0x80000000u }; // [56:32]|type=2
    u32x8 g1 = { 0x00010000u,                          // data_size=2B
                 64u << 16,                            // tensor_dim0 = 64
                 (2048u & 0xFFFFu) << 16,              // tensor_dim1 = 2048
                 64u << 16,                            // tile_dim0 = 64
                 32u,                                  // tile_dim1 = 32
                 64u,                                  // tensor_dim0_stride = 64
                 0u, 0u };
    asm volatile("tensor_load_to_lds %0, %1" :: "s"(g0), "s"(g1) : "memory");
}

// ---------------------------------------------------------------------------
// f32 -> f16 conversion
// ---------------------------------------------------------------------------
__global__ __launch_bounds__(256) void cvt_f32_f16_kernel(
    const float* __restrict__ src, _Float16* __restrict__ dst, int n)
{
    int i = blockIdx.x * blockDim.x + threadIdx.x;
    int stride = gridDim.x * blockDim.x;
    for (; i < n; i += stride) dst[i] = (_Float16)src[i];
}

// ---------------------------------------------------------------------------
// GEMM:  out = X(4096x1024) @ W(1024x1024) + bias
// mode 0: f16 output, head-split [B,H,L,D]   (Q/K/V projections)
// mode 1: f32 output, flat [4096,1024]       (final projection)
// Each wave owns FOUR 16-row M tiles sharing one B fragment.
// ---------------------------------------------------------------------------
__global__ __launch_bounds__(256) void gemm_wmma_kernel(
    const _Float16* __restrict__ X,
    const _Float16* __restrict__ W,
    const float*    __restrict__ bias,
    _Float16*       __restrict__ out_split,
    float*          __restrict__ out_flat,
    int mode)
{
    __shared__ alignas(32) _Float16 Bt[16][48];   // [n][k], padded row = 96B

    const int lane = threadIdx.x & 31;
    const int wave = threadIdx.x >> 5;
    const int n0   = blockIdx.x * 16;
    const int mg   = blockIdx.y * 8 + wave;       // 64-row M group (0..63)
    const int lrow = lane & 15;
    const int kbA  = (lane >> 4) * 16;
    const int kbB  = (lane >> 4) * 16;

    size_t xrow[4];
#pragma unroll
    for (int t = 0; t < 4; ++t)
        xrow[t] = (size_t)(mg * 64 + t * 16 + lrow) * kC;

    v8f acc[4] = {};

    for (int k0 = 0; k0 < kC; k0 += 32) {
        {
            int e  = threadIdx.x * 2;
            int kk = e >> 4;
            int nn = e & 15;
            v2h wv = *(const v2h*)(W + (size_t)(k0 + kk) * kC + n0 + nn);
            Bt[nn][kk]     = wv[0];
            Bt[nn + 1][kk] = wv[1];
        }
        __syncthreads();

        AFrag b;
        b.v = *(const v16h*)(&Bt[lrow][kbB]);
#pragma unroll
        for (int t = 0; t < 4; ++t) {
            AFrag a;
            a.h[0] = *(const v8h*)(X + xrow[t] + k0 + kbA);
            a.h[1] = *(const v8h*)(X + xrow[t] + k0 + kbA + 8);
            acc[t] = __builtin_amdgcn_wmma_f32_16x16x32_f16(
                false, a.v, false, b.v, (short)0, acc[t], false, false);
        }
        __syncthreads();
    }

    const int col  = n0 + lrow;
    const float bv = bias[col];

#pragma unroll
    for (int t = 0; t < 4; ++t) {
        const int mbase = mg * 64 + t * 16 + ((lane >> 4) << 3);
        if (mode == 0) {
            const int hh = col >> 6;
            const int dd = col & 63;
#pragma unroll
            for (int r = 0; r < 8; ++r) {
                int m    = mbase + r;
                int bidx = m >> 11;
                int l    = m & (kL - 1);
                out_split[(((size_t)(bidx * kH + hh) * kL) + l) * kD + dd] =
                    (_Float16)(acc[t][r] + bv);
            }
        } else {
#pragma unroll
            for (int r = 0; r < 8; ++r) {
                int m = mbase + r;
                out_flat[(size_t)m * kC + col] = acc[t][r] + bv;
            }
        }
    }
}

// ---------------------------------------------------------------------------
// Flash attention.  grid = (L/128, B*H), block = 256 (8 waves).
// K tile (32x64, 4KB): async global->LDS copy (each wave one 512B chunk).
// V tile (32x64, 4KB): TDM tensor_load_to_lds (wave 0), then LDS->LDS
// transpose into Vt.  Each wave: 16 query rows, online softmax, 8 WMMA/step.
// ---------------------------------------------------------------------------
__global__ __launch_bounds__(256) void attention_wmma_kernel(
    const _Float16* __restrict__ Qh,
    const _Float16* __restrict__ Kh,
    const _Float16* __restrict__ Vh,
    _Float16*       __restrict__ Y)
{
    __shared__ alignas(128) _Float16 Klds[32][64];     // raw [key][d], 4KB
    __shared__ alignas(128) _Float16 Vraw[32][64];     // raw [key][d], 4KB (TDM)
    __shared__ alignas(32)  _Float16 Vt[64][48];       // [d][key], padded
    __shared__ alignas(32)  _Float16 Pa[8][16][48];    // per-wave P, [row][k]

    const int lane = threadIdx.x & 31;
    const int wave = threadIdx.x >> 5;
    const int bh   = blockIdx.y;
    const int bi   = bh >> 4;
    const int hi   = bh & 15;

    const size_t base = (size_t)bh * kL * kD;
    const _Float16* Q = Qh + base;
    const _Float16* K = Kh + base;
    const _Float16* V = Vh + base;

    const int q0   = blockIdx.x * 128 + wave * 16;
    const int lrow = lane & 15;
    const int kbA  = (lane >> 4) * 16;
    const int kbB  = (lane >> 4) * 16;
    const int mb   = (lane >> 4) << 3;

    const unsigned klds_base = (unsigned)(uintptr_t)(&Klds[0][0]);
    const unsigned vraw_base = (unsigned)(uintptr_t)(&Vraw[0][0]);
    const unsigned chunk     = (unsigned)(wave * 512 + lane * 16); // bytes

    // Q fragments: two 16x32 A tiles covering d=0..63
    AFrag qa[2];
#pragma unroll
    for (int dc = 0; dc < 2; ++dc) {
        const _Float16* qp = Q + (size_t)(q0 + lrow) * kD + dc * 32;
        qa[dc].h[0] = *(const v8h*)(qp + kbA);
        qa[dc].h[1] = *(const v8h*)(qp + kbA + 8);
    }

    float m_run[8], l_run[8];
#pragma unroll
    for (int r = 0; r < 8; ++r) { m_run[r] = -1e30f; l_run[r] = 0.0f; }
    v8f o[4] = {};

    for (int key0 = 0; key0 < kL; key0 += 32) {
        // --- TDM stage of raw V tile (wave 0 issues the descriptor) --------
        if (wave == 0)
            tdm_load_v_tile_to_lds(vraw_base,
                (const char*)V + (size_t)key0 * kD * sizeof(_Float16));

        // --- async stage of K tile: each wave copies 512B ------------------
        async_load_b128_to_lds(klds_base + chunk,
            (const char*)(K + (size_t)key0 * kD) + chunk);

        if (key0 + 32 < kL) {
            __builtin_prefetch(K + (size_t)(key0 + 32) * kD, 0, 1);
            __builtin_prefetch(V + (size_t)(key0 + 32) * kD, 0, 1);
        }
        wait_asynccnt0();
        __builtin_amdgcn_s_wait_tensorcnt(0);
        __syncthreads();

        // --- transpose V tile: Vraw[key][d] -> Vt[d][key] (LDS->LDS) -------
        {
            int key  = threadIdx.x >> 3;           // 0..31
            int dblk = (threadIdx.x & 7) * 8;      // 0..56
            v8h vv = *(const v8h*)(&Vraw[key][dblk]);
#pragma unroll
            for (int j = 0; j < 8; ++j) Vt[dblk + j][key] = vv[j];
        }
        __syncthreads();

        // --- S = Q @ K^T (two 16x16 tiles over 32 keys), K from LDS --------
        v8f s0 = {}, s1 = {};
#pragma unroll
        for (int dc = 0; dc < 2; ++dc) {
            AFrag b0, b1;
            b0.v = *(const v16h*)(&Klds[lrow][dc * 32 + kbB]);
            b1.v = *(const v16h*)(&Klds[16 + lrow][dc * 32 + kbB]);
            s0 = __builtin_amdgcn_wmma_f32_16x16x32_f16(
                false, qa[dc].v, false, b0.v, (short)0, s0, false, false);
            s1 = __builtin_amdgcn_wmma_f32_16x16x32_f16(
                false, qa[dc].v, false, b1.v, (short)0, s1, false, false);
        }

        // --- online softmax (per lane: 8 rows of its half) -----------------
        const float scale = 0.125f;                // 1/sqrt(64)
        float p0[8], p1[8];
#pragma unroll
        for (int r = 0; r < 8; ++r) {
            float a = s0[r] * scale;
            float b = s1[r] * scale;
            float v = fmaxf(a, b);
#pragma unroll
            for (int off = 1; off < 16; off <<= 1)
                v = fmaxf(v, __shfl_xor(v, off, 16));
            float mnew  = fmaxf(m_run[r], v);
            float alpha = __expf(m_run[r] - mnew);
            p0[r] = __expf(a - mnew);
            p1[r] = __expf(b - mnew);
            float rs = p0[r] + p1[r];
#pragma unroll
            for (int off = 1; off < 16; off <<= 1)
                rs += __shfl_xor(rs, off, 16);
            l_run[r] = l_run[r] * alpha + rs;
            m_run[r] = mnew;
#pragma unroll
            for (int dt = 0; dt < 4; ++dt) o[dt][r] *= alpha;
        }

        // --- P: C/D layout -> A layout via per-wave LDS --------------------
#pragma unroll
        for (int r = 0; r < 8; ++r) {
            Pa[wave][mb + r][lrow]      = (_Float16)p0[r];
            Pa[wave][mb + r][16 + lrow] = (_Float16)p1[r];
        }
        AFrag pa;
        pa.h[0] = *(const v8h*)(&Pa[wave][lrow][kbA]);
        pa.h[1] = *(const v8h*)(&Pa[wave][lrow][kbA + 8]);

        // --- O += P @ V ----------------------------------------------------
#pragma unroll
        for (int dt = 0; dt < 4; ++dt) {
            AFrag bv;
            bv.v = *(const v16h*)(&Vt[dt * 16 + lrow][kbB]);
            o[dt] = __builtin_amdgcn_wmma_f32_16x16x32_f16(
                false, pa.v, false, bv.v, (short)0, o[dt], false, false);
        }

        __syncthreads();
    }

    // --- epilogue: normalize, merge heads into Y[B,L,C] (f16) --------------
#pragma unroll
    for (int r = 0; r < 8; ++r) {
        float inv = 1.0f / l_run[r];
        int m = q0 + mb + r;
        size_t yoff = ((size_t)(bi * kL + m)) * kC + hi * kD;
#pragma unroll
        for (int dt = 0; dt < 4; ++dt)
            Y[yoff + dt * 16 + lrow] = (_Float16)(o[dt][r] * inv);
    }
}

// ---------------------------------------------------------------------------
// Host-side launcher
// ---------------------------------------------------------------------------
extern "C" void kernel_launch(void* const* d_in, const int* in_sizes, int n_in,
                              void* d_out, int out_size, void* d_ws, size_t ws_size,
                              hipStream_t stream) {
    const float* q  = (const float*)d_in[0];
    const float* k  = (const float*)d_in[1];
    const float* v  = (const float*)d_in[2];
    const float* Wq = (const float*)d_in[3];
    const float* bq = (const float*)d_in[4];
    const float* Wk = (const float*)d_in[5];
    const float* bk = (const float*)d_in[6];
    const float* Wv = (const float*)d_in[7];
    const float* bv = (const float*)d_in[8];
    const float* Wo = (const float*)d_in[9];
    const float* bo = (const float*)d_in[10];
    float* out = (float*)d_out;

    const size_t M4 = (size_t)kNT * kC;

    _Float16* ws   = (_Float16*)d_ws;
    _Float16* Xq   = ws;
    _Float16* Xk   = Xq + M4;
    _Float16* Xv   = Xk + M4;
    _Float16* Wq16 = Xv + M4;
    _Float16* Wk16 = Wq16 + M4;
    _Float16* Wv16 = Wk16 + M4;
    _Float16* Wo16 = Wv16 + M4;
    _Float16* Qh   = Wo16 + M4;
    _Float16* Kh   = Qh + M4;
    _Float16* Vh   = Kh + M4;
    _Float16* Yb   = Vh + M4;

    const int n4 = (int)M4;
    cvt_f32_f16_kernel<<<1024, 256, 0, stream>>>(q,  Xq,   n4);
    cvt_f32_f16_kernel<<<1024, 256, 0, stream>>>(k,  Xk,   n4);
    cvt_f32_f16_kernel<<<1024, 256, 0, stream>>>(v,  Xv,   n4);
    cvt_f32_f16_kernel<<<1024, 256, 0, stream>>>(Wq, Wq16, n4);
    cvt_f32_f16_kernel<<<1024, 256, 0, stream>>>(Wk, Wk16, n4);
    cvt_f32_f16_kernel<<<1024, 256, 0, stream>>>(Wv, Wv16, n4);
    cvt_f32_f16_kernel<<<1024, 256, 0, stream>>>(Wo, Wo16, n4);

    dim3 gblock(256);
    dim3 ggrid(kC / 16, kNT / (64 * 8));  // (64, 8)
    gemm_wmma_kernel<<<ggrid, gblock, 0, stream>>>(Xq, Wq16, bq, Qh, nullptr, 0);
    gemm_wmma_kernel<<<ggrid, gblock, 0, stream>>>(Xk, Wk16, bk, Kh, nullptr, 0);
    gemm_wmma_kernel<<<ggrid, gblock, 0, stream>>>(Xv, Wv16, bv, Vh, nullptr, 0);

    dim3 agrid(kL / 128, kB * kH);        // (16, 32)
    attention_wmma_kernel<<<agrid, gblock, 0, stream>>>(Qh, Kh, Vh, Yb);

    gemm_wmma_kernel<<<ggrid, gblock, 0, stream>>>(Yb, Wo16, bo, nullptr, out, 1);
}
